// LoicGCN_16277926052611
// MI455X (gfx1250) — matrix operand order repaired and checked by reading, not compiled, not run on hardware
//
#include <hip/hip_runtime.h>
#include <hip/hip_bf16.h>

typedef _Float16 v16h __attribute__((ext_vector_type(16)));
typedef _Float16 v8h  __attribute__((ext_vector_type(8)));
typedef float    v8f  __attribute__((ext_vector_type(8)));

// ---------------- sparse phase ----------------

__global__ void zero_f32(float* p, long n) {
    long i = (long)blockIdx.x * blockDim.x + threadIdx.x;
    if (i < n) p[i] = 0.0f;
}

__global__ void deg_kernel(const int* __restrict__ src, const float* __restrict__ ew,
                           float* __restrict__ deg, int E) {
    int e = blockIdx.x * blockDim.x + threadIdx.x;
    if (e >= E) return;
    atomicAdd(&deg[src[e]], ew[e]);
}

__global__ void dinv_kernel(const float* __restrict__ deg, float* __restrict__ dinv, int n) {
    int i = blockIdx.x * blockDim.x + threadIdx.x;
    if (i >= n) return;
    float d = deg[i];
    dinv[i] = d > 0.0f ? __frsqrt_rn(d) : 0.0f;
}

__global__ void norm_kernel(const int* __restrict__ src, const int* __restrict__ dst,
                            const float* __restrict__ ew, const float* __restrict__ dinv,
                            float* __restrict__ norm, int E) {
    int e = blockIdx.x * blockDim.x + threadIdx.x;
    if (e >= E) return;
    norm[e] = -ew[e] * dinv[src[e]] * dinv[dst[e]];
}

// tx1[dst] += norm[e] * x[src]   (F features per edge; L2-resident atomics)
template <int F>
__global__ void scatter_kernel(const int* __restrict__ src, const int* __restrict__ dst,
                               const float* __restrict__ norm, const float* __restrict__ x,
                               float* __restrict__ tx1, int E) {
    int e = blockIdx.x * blockDim.x + threadIdx.x;
    if (e >= E) return;
    int s = src[e], d = dst[e];
    float nm = norm[e];
    const float* xr = x + (size_t)s * F;
    float* tr = tx1 + (size_t)d * F;
#pragma unroll 4
    for (int f = 0; f < F; ++f) atomicAdd(&tr[f], nm * xr[f]);
}

// ---------------- packing / conversion ----------------

// A = [ x (Fin cols, zero-padded to Kpad) | tx1 (Fin cols, padded to Kpad) ]  as f16
__global__ void packA_kernel(const float* __restrict__ x, const float* __restrict__ tx1,
                             _Float16* __restrict__ A, int nrows, int Fin, int Kpad) {
    int W = 2 * Kpad;
    long i = (long)blockIdx.x * blockDim.x + threadIdx.x;
    if (i >= (long)nrows * W) return;
    int col = (int)(i % W);
    long row = i / W;
    float v = 0.0f;
    if (col < Kpad) {
        if (col < Fin) v = x[row * Fin + col];
    } else {
        int c = col - Kpad;
        if (c < Fin) v = tx1[row * Fin + c];
    }
    A[i] = (_Float16)v;
}

// Pack [W0 ; zeros ; W1 ; zeros] (logical [K=2*Kpad, Fout]) directly into WMMA
// B-fragment layout: Wfrag[((tn*KT + t)*32 + lane)*16 + j] = Wlogical[t*32 + kb(lane) + j, tn*16 + (lane&15)]
// where kb(lane) = (lane>=16) ? 16 : 0.  One aligned 32B load per lane per k-chunk in the GEMM.
__global__ void packWfrag_kernel(const float* __restrict__ W0, const float* __restrict__ W1,
                                 _Float16* __restrict__ Wfrag, int Fin, int Fout, int Kpad) {
    int KT = (2 * Kpad) / 32;
    int tilesN = Fout / 16;
    int total = tilesN * KT * 32 * 16;
    int i = blockIdx.x * blockDim.x + threadIdx.x;
    if (i >= total) return;
    int j    = i & 15;
    int lane = (i >> 4) & 31;
    int rest = i >> 9;
    int t    = rest % KT;
    int tn   = rest / KT;
    int k    = t * 32 + ((lane >= 16) ? 16 : 0) + j;
    int col  = tn * 16 + (lane & 15);
    float v = 0.0f;
    if (k < Fin)                               v = W0[k * Fout + col];
    else if (k >= Kpad && k < Kpad + Fin)      v = W1[(k - Kpad) * Fout + col];
    Wfrag[i] = (_Float16)v;
}

__global__ void cvt_f16(const float* __restrict__ in, _Float16* __restrict__ out, long n) {
    long i = (long)blockIdx.x * blockDim.x + threadIdx.x;
    if (i < n) out[i] = (_Float16)in[i];
}

// ---------------- ChebConv GEMM: O = relu(A[M, KT*32] @ Wfrag + bias) ----------------
// Wfrag pre-packed in B-fragment layout. Each wave handles MPW M-tiles of one N-tile,
// holding all KT B-fragments in registers.
template <bool RELU, int KT, int NOUT, int MPW>
__global__ void gemm_cheb(const _Float16* __restrict__ A, const _Float16* __restrict__ Wfrag,
                          const float* __restrict__ bias, float* __restrict__ O, int Mrows) {
    constexpr int K = KT * 32;
    constexpr int tilesN = NOUT / 16;
    int lane = threadIdx.x & 31;
    int wave = (blockIdx.x * blockDim.x + threadIdx.x) >> 5;
    int groups = (Mrows >> 4) / MPW;
    if (wave >= groups * tilesN) return;           // wave-uniform: EXEC all-ones inside
    int tn = wave % tilesN;
    int g  = wave / tilesN;
    int hl = lane & 15;
    bool hi = lane >= 16;
    int kbA = hi ? 16 : 0;
    int col = tn * 16 + hl;

    // B fragments: one aligned 32B load per k-chunk
    v16h b[KT];
    const _Float16* bp = Wfrag + ((size_t)(tn * KT) * 32 + lane) * 16;
#pragma unroll
    for (int t = 0; t < KT; ++t) b[t] = *(const v16h*)(bp + (size_t)t * 512);

    float bv = bias[col];
#pragma unroll
    for (int mi = 0; mi < MPW; ++mi) {
        int tm = g * MPW + mi;
        const _Float16* arow = A + (size_t)(tm * 16 + hl) * K + kbA;
        v8f c = {};
#pragma unroll
        for (int t = 0; t < KT; ++t) {
            v16h a;
            *(v8h*)&a       = *(const v8h*)(arow + t * 32);
            *((v8h*)&a + 1) = *(const v8h*)(arow + t * 32 + 8);
            c = __builtin_amdgcn_wmma_f32_16x16x32_f16(false, a, false, b[t], (short)0, c, false, false);
        }
        float* orow = O + (size_t)(tm * 16 + (hi ? 8 : 0)) * NOUT + col;
#pragma unroll
        for (int r = 0; r < 8; ++r) {
            float v = c[r] + bv;
            if (RELU) v = fmaxf(v, 0.0f);
            orow[(size_t)r * NOUT] = v;           // compile-time immediate offsets
        }
    }
}

// ---------------- FC GEMM: O = A[M,K] @ B^T + bias, B stored [NOUT, K] (torch Linear) ----------------
template <int NOUT>
__global__ void gemm_fc(const _Float16* __restrict__ A, const _Float16* __restrict__ B,
                        const float* __restrict__ bias, float* __restrict__ O,
                        int Mrows, int K) {
    constexpr int tilesN = NOUT / 16;
    int lane = threadIdx.x & 31;
    int wave = (blockIdx.x * blockDim.x + threadIdx.x) >> 5;
    int totalTiles = (Mrows >> 4) * tilesN;
    if (wave >= totalTiles) return;
    int tn = wave % tilesN;
    int tm = wave / tilesN;
    int hl = lane & 15;
    bool hi = lane >= 16;
    int kbA = hi ? 16 : 0;
    int kbB = hi ? 16 : 0;
    int col = tn * 16 + hl;

    const _Float16* arow = A + (size_t)(tm * 16 + hl) * K + kbA;
    const _Float16* brow = B + (size_t)col * K + kbB;

    v8f c = {};
#pragma unroll 4
    for (int kc = 0; kc < K; kc += 32) {
        v16h a;
        *(v8h*)&a       = *(const v8h*)(arow + kc);
        *((v8h*)&a + 1) = *(const v8h*)(arow + kc + 8);
        v16h b;
        *(v8h*)&b       = *(const v8h*)(brow + kc);
        *((v8h*)&b + 1) = *(const v8h*)(brow + kc + 8);
        c = __builtin_amdgcn_wmma_f32_16x16x32_f16(false, a, false, b, (short)0, c, false, false);
    }

    float bv = bias[col];
    float* orow = O + (size_t)(tm * 16 + (hi ? 8 : 0)) * NOUT + col;
#pragma unroll
    for (int r = 0; r < 8; ++r) orow[(size_t)r * NOUT] = c[r] + bv;
}

// ---------------- tiny fc3 (Nout=2) ----------------
__global__ void fc3_kernel(const float* __restrict__ h, const float* __restrict__ w,
                           const float* __restrict__ b, float* __restrict__ out, int Bsz) {
    int i = blockIdx.x * blockDim.x + threadIdx.x;
    if (i >= Bsz * 2) return;
    int g = i >> 1, o = i & 1;
    const float* hr = h + (size_t)g * 128;
    const float* wr = w + (size_t)o * 128;
    float acc = b[o];
#pragma unroll 8
    for (int k = 0; k < 128; ++k) acc = fmaf(hr[k], wr[k], acc);
    out[i] = acc;
}

// ---------------- driver ----------------

static inline int cdiv(long a, int b) { return (int)((a + b - 1) / b); }

extern "C" void kernel_launch(void* const* d_in, const int* in_sizes, int n_in,
                              void* d_out, int out_size, void* d_ws, size_t ws_size,
                              hipStream_t stream) {
    const float* x     = (const float*)d_in[0];
    const int*   ei    = (const int*)d_in[1];
    const float* ew    = (const float*)d_in[2];
    const float* W1_0  = (const float*)d_in[3];
    const float* W1_1  = (const float*)d_in[4];
    const float* b1    = (const float*)d_in[5];
    const float* W2_0  = (const float*)d_in[6];
    const float* W2_1  = (const float*)d_in[7];
    const float* b2    = (const float*)d_in[8];
    const float* W3_0  = (const float*)d_in[9];
    const float* W3_1  = (const float*)d_in[10];
    const float* b3    = (const float*)d_in[11];
    const float* fc1w  = (const float*)d_in[12];
    const float* fc1b  = (const float*)d_in[13];
    const float* fc2w  = (const float*)d_in[14];
    const float* fc2b  = (const float*)d_in[15];
    const float* fc3w  = (const float*)d_in[16];
    const float* fc3b  = (const float*)d_in[17];

    const long N = in_sizes[0] / 50;          // 262144
    const int  E = in_sizes[2];               // 4194304
    const int  Bsz = (int)(N / 512);          // 512 graphs
    const int* src = ei;
    const int* dst = ei + E;

    // workspace layout (256B aligned)
    char* ws = (char*)d_ws;
    size_t off = 0;
    auto alloc = [&](size_t bytes) { char* p = ws + off; off = (off + bytes + 255) & ~(size_t)255; return p; };
    float*    deg   = (float*)   alloc(N * 4);
    float*    dinv  = (float*)   alloc(N * 4);
    float*    norm  = (float*)   alloc((size_t)E * 4);
    float*    tx1   = (float*)   alloc(N * 64 * 4);          // reused each layer
    _Float16* Abuf  = (_Float16*)alloc(N * 128 * 2);         // reused each layer
    float*    h1    = (float*)   alloc(N * 32 * 4);
    float*    h2    = (float*)   alloc(N * 32 * 4);
    float*    h3    = (float*)   alloc(N * 16 * 4);
    _Float16* h3h   = (_Float16*)alloc(N * 16 * 2);
    _Float16* Wfrag = (_Float16*)alloc(128 * 32 * 2);        // reused per layer (max K*Fout)
    _Float16* fc1wh = (_Float16*)alloc((size_t)256 * 8192 * 2);
    _Float16* fc2wh = (_Float16*)alloc((size_t)128 * 256 * 2);
    float*    fc1o  = (float*)   alloc((size_t)Bsz * 256 * 4);
    _Float16* fc1oh = (_Float16*)alloc((size_t)Bsz * 256 * 2);
    float*    fc2o  = (float*)   alloc((size_t)Bsz * 128 * 4);

    const int T = 256;

    // ---- normalization ----
    zero_f32<<<cdiv(N, T), T, 0, stream>>>(deg, N);
    deg_kernel<<<cdiv(E, T), T, 0, stream>>>(src, ew, deg, E);
    dinv_kernel<<<cdiv(N, T), T, 0, stream>>>(deg, dinv, (int)N);
    norm_kernel<<<cdiv(E, T), T, 0, stream>>>(src, dst, ew, dinv, norm, E);

    // ---- ChebConv layer 1: 50 -> 32 (Kpad=64, K=128, KT=4) ----
    zero_f32<<<cdiv(N * 50, T), T, 0, stream>>>(tx1, N * 50);
    scatter_kernel<50><<<cdiv(E, T), T, 0, stream>>>(src, dst, norm, x, tx1, E);
    packA_kernel<<<cdiv(N * 128, T), T, 0, stream>>>(x, tx1, Abuf, (int)N, 50, 64);
    packWfrag_kernel<<<cdiv(128 * 32, T), T, 0, stream>>>(W1_0, W1_1, Wfrag, 50, 32, 64);
    {
        int waves = (int)((N / 16) / 4) * 2;             // groups * tilesN
        gemm_cheb<true, 4, 32, 4><<<cdiv((long)waves * 32, T), T, 0, stream>>>(Abuf, Wfrag, b1, h1, (int)N);
    }

    // ---- ChebConv layer 2: 32 -> 32 (Kpad=32, K=64, KT=2) ----
    zero_f32<<<cdiv(N * 32, T), T, 0, stream>>>(tx1, N * 32);
    scatter_kernel<32><<<cdiv(E, T), T, 0, stream>>>(src, dst, norm, h1, tx1, E);
    packA_kernel<<<cdiv(N * 64, T), T, 0, stream>>>(h1, tx1, Abuf, (int)N, 32, 32);
    packWfrag_kernel<<<cdiv(64 * 32, T), T, 0, stream>>>(W2_0, W2_1, Wfrag, 32, 32, 32);
    {
        int waves = (int)((N / 16) / 4) * 2;
        gemm_cheb<true, 2, 32, 4><<<cdiv((long)waves * 32, T), T, 0, stream>>>(Abuf, Wfrag, b2, h2, (int)N);
    }

    // ---- ChebConv layer 3: 32 -> 16 (Kpad=32, K=64, KT=2) ----
    zero_f32<<<cdiv(N * 32, T), T, 0, stream>>>(tx1, N * 32);
    scatter_kernel<32><<<cdiv(E, T), T, 0, stream>>>(src, dst, norm, h2, tx1, E);
    packA_kernel<<<cdiv(N * 64, T), T, 0, stream>>>(h2, tx1, Abuf, (int)N, 32, 32);
    packWfrag_kernel<<<cdiv(64 * 16, T), T, 0, stream>>>(W3_0, W3_1, Wfrag, 32, 16, 32);
    {
        int waves = (int)((N / 16) / 4) * 1;
        gemm_cheb<true, 2, 16, 4><<<cdiv((long)waves * 32, T), T, 0, stream>>>(Abuf, Wfrag, b3, h3, (int)N);
    }

    // ---- FC head ----
    cvt_f16<<<cdiv(N * 16, T), T, 0, stream>>>(h3, h3h, N * 16);                 // viewed [512, 8192]
    cvt_f16<<<cdiv((long)256 * 8192, T), T, 0, stream>>>(fc1w, fc1wh, (long)256 * 8192);
    cvt_f16<<<cdiv((long)128 * 256, T), T, 0, stream>>>(fc2w, fc2wh, (long)128 * 256);

    {   // fc1: [512,8192] @ [8192,256]
        int tiles = (Bsz / 16) * (256 / 16);
        gemm_fc<256><<<cdiv((long)tiles * 32, T), T, 0, stream>>>(h3h, fc1wh, fc1b, fc1o, Bsz, 8192);
    }
    cvt_f16<<<cdiv((long)Bsz * 256, T), T, 0, stream>>>(fc1o, fc1oh, (long)Bsz * 256);
    {   // fc2: [512,256] @ [256,128]
        int tiles = (Bsz / 16) * (128 / 16);
        gemm_fc<128><<<cdiv((long)tiles * 32, T), T, 0, stream>>>(fc1oh, fc2wh, fc2b, fc2o, Bsz, 256);
    }
    fc3_kernel<<<cdiv(Bsz * 2, T), T, 0, stream>>>(fc2o, fc3w, fc3b, (float*)d_out, Bsz);
}
